// ContextVectorNN_52759378264564
// MI455X (gfx1250) — matrix operand order repaired and checked by reading, not compiled, not run on hardware
//
#include <hip/hip_runtime.h>

typedef __attribute__((ext_vector_type(16))) __bf16        v16bf;
typedef __attribute__((ext_vector_type(8)))  float         v8f;
typedef __attribute__((ext_vector_type(8)))  unsigned int  v8u;
typedef int v4i_vec __attribute__((vector_size(16)));   // matches async-lds builtin param
typedef __attribute__((address_space(3))) v4i_vec* lds_v4i_ptr;

#define B_  32
#define L_  4096
#define FT_ 512
#define FS_ 512
#define H_  512
#define C_  1025   // FT + FS + 1

// workspace layout (byte offsets)
#define WS_W1BF   0u          // 512*512 ushort  = 524288 B
#define WS_UBIAS  524288u     // 32*512 float    =  65536 B
#define WS_LOGIT  589824u     // 32*4096 float   = 524288 B  (total ~1.07 MB)

// gfx1250 async global->LDS copy path (ASYNCcnt); fall back to manual copy
#if __has_builtin(__builtin_amdgcn_global_load_async_to_lds_b128) && \
    __has_builtin(__builtin_amdgcn_s_wait_asynccnt)
#define USE_ASYNC_LDS 1
#define ASYNC_CP16(dst, src)                                               \
  __builtin_amdgcn_global_load_async_to_lds_b128(                          \
      (v4i_vec*)(src), (lds_v4i_ptr)(dst), 0, 0)
#else
#define USE_ASYNC_LDS 0
#endif

__device__ __forceinline__ unsigned int bf16_1(float f) {
  unsigned int u = __float_as_uint(f);
  u += 0x7FFFu + ((u >> 16) & 1u);       // round-to-nearest-even
  return u >> 16;
}
__device__ __forceinline__ unsigned int bfpack2(float lo, float hi) {
  return (bf16_1(lo) & 0xFFFFu) | (bf16_1(hi) << 16);
}
__device__ __forceinline__ float fast_tanh(float x) {
  float e = __expf(2.0f * x);            // v_exp_f32 (TRANS)
  return 1.0f - 2.0f * __builtin_amdgcn_rcpf(e + 1.0f);
}

// ---------------- prep kernels ----------------

__global__ void k_prep_w1(const float* __restrict__ W1, unsigned short* __restrict__ w1bf) {
  int idx = blockIdx.x * 256 + threadIdx.x;      // 0 .. 262143
  int h = idx >> 9, c = idx & 511;
  w1bf[idx] = (unsigned short)bf16_1(W1[h * C_ + c]);
}

__global__ void k_prep_ubias(const float* __restrict__ W1, const float* __restrict__ b1,
                             const float* __restrict__ summary, float* __restrict__ ubias) {
  int h = blockIdx.x * 256 + threadIdx.x;        // 0..511
  int b = blockIdx.y;
  const float* wrow = W1 + h * C_ + FT_;
  const float* s    = summary + b * FS_;
  float acc = b1[h];
  #pragma unroll 8
  for (int i = 0; i < FS_; ++i) acc += wrow[i] * s[i];
  ubias[b * H_ + h] = acc;
}

__global__ void k_zero_ctx(float* __restrict__ ctx) {
  ctx[blockIdx.x * 256 + threadIdx.x] = 0.0f;
}

// ---------------- main fused GEMM + tanh + w2-reduction kernel ----------------
// grid (L/128, B), 256 threads = 8 waves; wave owns 16 columns of L.
// B-fragments (ts columns, bf16) persist in registers; A (W1 bf16 rows) is
// double-buffered through LDS per 16-row H chunk, copied with the gfx1250
// async-to-LDS engine when available.

__global__ __launch_bounds__(256) void k_attn_logits(
    const float* __restrict__ text, const float* __restrict__ cov,
    const float* __restrict__ W1,   const float* __restrict__ W2,
    const float* __restrict__ b2p,
    const unsigned short* __restrict__ w1bf,
    const float* __restrict__ ubias,
    float* __restrict__ logits)
{
  __shared__ __align__(16) unsigned short sA[2][16 * 512];  // 2 x 16KB A tiles
  __shared__ float s_u[H_];
  __shared__ float s_w1c[H_];
  __shared__ float s_w2[H_];
  __shared__ float s_cov[128];

  const int tid  = threadIdx.x;
  const int b    = blockIdx.y;
  const int l0   = blockIdx.x * 128;
  const int wave = tid >> 5;
  const int lane = tid & 31;

  for (int i = tid; i < H_; i += 256) {
    s_u[i]   = ubias[b * H_ + i];
    s_w1c[i] = W1[i * C_ + (C_ - 1)];   // coverage column of W1
    s_w2[i]  = W2[i];
  }
  if (tid < 128) s_cov[tid] = cov[b * L_ + l0 + tid];

  // ---- load B fragments: wave's 16 columns, all K=512, bf16 ----
  // B(32x16) layout: lane n<16 -> col n, K 0..15 (pairs per VGPR); lane n+16 -> K 16..31
  const int   col   = l0 + wave * 16 + (lane & 15);
  const int   khalf = (lane >> 4) * 16;
  const float* tbase = text + (size_t)(b * L_ + col) * FT_;
  v16bf bfrag[16];
  #pragma unroll
  for (int kc = 0; kc < 16; ++kc) {
    const float4* p = (const float4*)(tbase + kc * 32 + khalf);
    float4 f0 = p[0], f1 = p[1], f2 = p[2], f3 = p[3];
    v8u u;
    u[0] = bfpack2(f0.x, f0.y); u[1] = bfpack2(f0.z, f0.w);
    u[2] = bfpack2(f1.x, f1.y); u[3] = bfpack2(f1.z, f1.w);
    u[4] = bfpack2(f2.x, f2.y); u[5] = bfpack2(f2.z, f2.w);
    u[6] = bfpack2(f3.x, f3.y); u[7] = bfpack2(f3.z, f3.w);
    bfrag[kc] = __builtin_bit_cast(v16bf, u);
  }

  // preload A chunk 0 (16 rows x 512 K bf16 = 16KB, cooperative)
#if USE_ASYNC_LDS
  {
    const unsigned short* src = w1bf;
    unsigned short*       dst = &sA[0][0];
    #pragma unroll
    for (int i = 0; i < 4; ++i) {
      const int e = (tid + i * 256) * 8;   // 8 ushorts = 16 bytes
      ASYNC_CP16(dst + e, src + e);
    }
    __builtin_amdgcn_s_wait_asynccnt(0);
  }
#else
  {
    const uint4* src = (const uint4*)w1bf;
    uint4*       dst = (uint4*)&sA[0][0];
    #pragma unroll
    for (int i = 0; i < 4; ++i) dst[tid + i * 256] = src[tid + i * 256];
  }
#endif
  __syncthreads();

  const float covv = s_cov[wave * 16 + (lane & 15)];
  const int   arow = lane & 15;
  const int   asub = (lane >> 4) * 16;
  float logit_acc = 0.0f;

  for (int hc = 0; hc < 32; ++hc) {
    const int cur = hc & 1;
    if (hc + 1 < 32) {  // prefetch next A tile into the other buffer
#if USE_ASYNC_LDS
      const unsigned short* src = w1bf + (size_t)(hc + 1) * 16 * 512;
      unsigned short*       dst = &sA[cur ^ 1][0];
      #pragma unroll
      for (int i = 0; i < 4; ++i) {
        const int e = (tid + i * 256) * 8;
        ASYNC_CP16(dst + e, src + e);
      }
#else
      const uint4* src = (const uint4*)(w1bf + (size_t)(hc + 1) * 16 * 512);
      uint4*       dst = (uint4*)&sA[cur ^ 1][0];
      #pragma unroll
      for (int i = 0; i < 4; ++i) dst[tid + i * 256] = src[tid + i * 256];
#endif
    }

    v8f acc = {};
    const unsigned short* pr = &sA[cur][arow * 512];
    #pragma unroll
    for (int kc = 0; kc < 16; ++kc) {
      // A(16x32) layout: lane<16 holds K 0..15, lane>=16 holds K 16..31
      uint4 lo = *(const uint4*)(pr + kc * 32 + asub);
      uint4 hi = *(const uint4*)(pr + kc * 32 + 8 + asub);
      v8u au;
      au[0] = lo.x; au[1] = lo.y; au[2] = lo.z; au[3] = lo.w;
      au[4] = hi.x; au[5] = hi.y; au[6] = hi.z; au[7] = hi.w;
      v16bf a = __builtin_bit_cast(v16bf, au);
      acc = __builtin_amdgcn_wmma_f32_16x16x32_bf16(false, a, false, bfrag[kc],
                                                    (short)0, acc, false, false);
    }

    // fused epilogue: bias + coverage rank-1 + tanh + dot with w2
    const int hbase = hc * 16 + ((lane >> 4) ? 8 : 0);
    #pragma unroll
    for (int r = 0; r < 8; ++r) {
      const int h = hbase + r;
      float z = acc[r] + s_u[h] + s_w1c[h] * covv;
      logit_acc += s_w2[h] * fast_tanh(z);
    }
#if USE_ASYNC_LDS
    __builtin_amdgcn_s_wait_asynccnt(0);   // LDS writes visible before barrier
#endif
    __syncthreads();
  }

  // lanes n and n+16 cover disjoint halves of H for column (n&15)
  float total = logit_acc + __shfl_xor(logit_acc, 16, 32) + b2p[0];
  if (lane < 16) logits[b * L_ + l0 + wave * 16 + lane] = total;
}

// ---------------- masked softmax over L, per batch ----------------

__global__ __launch_bounds__(256) void k_softmax(
    const float* __restrict__ logits, const int* __restrict__ tlen,
    float* __restrict__ att)
{
  __shared__ float red[256];
  const int b = blockIdx.x, tid = threadIdx.x;
  int n = tlen[b]; if (n < 1) n = 1; if (n > L_) n = L_;
  const float* lg = logits + b * L_;
  float mx = -3.402823466e38f;
  for (int l = tid; l < n; l += 256) mx = fmaxf(mx, lg[l]);
  red[tid] = mx; __syncthreads();
  for (int s = 128; s > 0; s >>= 1) {
    if (tid < s) red[tid] = fmaxf(red[tid], red[tid + s]);
    __syncthreads();
  }
  mx = red[0]; __syncthreads();

  float sum = 0.0f;
  float* ob = att + b * L_;
  for (int l = tid; l < L_; l += 256) {
    float e = (l < n) ? __expf(lg[l] - mx) : 0.0f;
    ob[l] = e; sum += e;
  }
  red[tid] = sum; __syncthreads();
  for (int s = 128; s > 0; s >>= 1) {
    if (tid < s) red[tid] += red[tid + s];
    __syncthreads();
  }
  const float inv = 1.0f / red[0];
  for (int l = tid; l < L_; l += 256) ob[l] *= inv;
}

// ---------------- context = sum_l att[l] * ts[:, l] ----------------
// grid (FT/128, L/512, B); each block reduces 512 l-steps for 128 features.

__global__ __launch_bounds__(128) void k_context(
    const float* __restrict__ text, const float* __restrict__ att,
    float* __restrict__ ctx)
{
  const int f  = blockIdx.x * 128 + threadIdx.x;
  const int l0 = blockIdx.y * 512;
  const int b  = blockIdx.z;
  const float* ab = att + b * L_ + l0;
  const float* tb = text + (size_t)(b * L_ + l0) * FT_ + f;
  float acc = 0.0f;
  #pragma unroll 4
  for (int l = 0; l < 512; ++l) acc += ab[l] * tb[(size_t)l * FT_];
  atomicAdd(&ctx[b * FT_ + f], acc);
}

// ---------------- launcher ----------------

extern "C" void kernel_launch(void* const* d_in, const int* in_sizes, int n_in,
                              void* d_out, int out_size, void* d_ws, size_t ws_size,
                              hipStream_t stream) {
  const float* text    = (const float*)d_in[0];   // [B,L,FT]
  const float* summary = (const float*)d_in[1];   // [B,FS]
  const float* cov     = (const float*)d_in[2];   // [B,1,L]
  const float* W1      = (const float*)d_in[3];   // [H,C]
  const float* b1      = (const float*)d_in[4];   // [H]
  const float* W2      = (const float*)d_in[5];   // [1,H]
  const float* b2      = (const float*)d_in[6];   // [1]
  const int*   tlen    = (const int*)d_in[7];     // [B]

  float* out = (float*)d_out;
  float* ctx = out;                 // [B,FT]   = 16384 floats
  float* att = out + B_ * FT_;      // [B,1,L]  = 131072 floats

  char* ws = (char*)d_ws;
  unsigned short* w1bf  = (unsigned short*)(ws + WS_W1BF);
  float*          ubias = (float*)(ws + WS_UBIAS);
  float*          logit = (float*)(ws + WS_LOGIT);

  k_prep_w1   <<<1024, 256, 0, stream>>>(W1, w1bf);
  k_prep_ubias<<<dim3(2, B_), 256, 0, stream>>>(W1, b1, summary, ubias);
  k_zero_ctx  <<<64, 256, 0, stream>>>(ctx);
  k_attn_logits<<<dim3(L_ / 128, B_), 256, 0, stream>>>(text, cov, W1, W2, b2,
                                                        w1bf, ubias, logit);
  k_softmax   <<<B_, 256, 0, stream>>>(logit, tlen, att);
  k_context   <<<dim3(FT_ / 128, L_ / 512, B_), 128, 0, stream>>>(text, att, ctx);
}
